// HGraphConvolution_40295383171110
// MI455X (gfx1250) — matrix operand order repaired and checked by reading, not compiled, not run on hardware
//
#include <hip/hip_runtime.h>
#include <hip/hip_bf16.h>

#define MIN_NORM 1e-15f
#define MAXNORM  (1.0f - 4e-3f)   // (1 - BALL_EPS)/sqrt(c), c = 1

typedef __attribute__((ext_vector_type(16))) _Float16 v16h;
typedef __attribute__((ext_vector_type(8)))  _Float16 v8h;
typedef __attribute__((ext_vector_type(8)))  float    v8f;

// ---------------- wave32 reduction ----------------
__device__ __forceinline__ float wsum(float v) {
#pragma unroll
  for (int m = 16; m >= 1; m >>= 1) v += __shfl_xor(v, m, 32);
  return v;
}

// ---------------- utility kernels ----------------
__global__ void zero_kernel(float* __restrict__ p, long long n) {
  long long i = (long long)blockIdx.x * blockDim.x + threadIdx.x;
  if (i < n) p[i] = 0.0f;
}

__global__ void cvt_f16_kernel(const float* __restrict__ s, _Float16* __restrict__ d, long long n) {
  long long i = (long long)blockIdx.x * blockDim.x + threadIdx.x;
  if (i < n) d[i] = (_Float16)s[i];
}

__global__ void concat_kernel(const float* __restrict__ a, int Da,
                              const float* __restrict__ b, int Db,
                              float* __restrict__ out, long long N) {
  int Dt = Da + Db;
  long long i = (long long)blockIdx.x * blockDim.x + threadIdx.x;
  if (i >= N * (long long)Dt) return;
  long long row = i / Dt;
  int col = (int)(i - row * Dt);
  out[i] = (col < Da) ? a[row * Da + col] : b[row * Db + (col - Da)];
}

__global__ void copyx_kernel(const float* __restrict__ x, float* __restrict__ out, long long N) {
  long long i = (long long)blockIdx.x * blockDim.x + threadIdx.x;
  if (i >= N * 128LL) return;
  long long row = i >> 7;
  int col = (int)(i & 127);
  out[row * 384 + 256 + col] = x[i];
}

// ---------------- graph aggregation ----------------
__global__ void degree_kernel(const int* __restrict__ dst, long long E, float* __restrict__ deg) {
  long long e = (long long)blockIdx.x * blockDim.x + threadIdx.x;
  if (e < E) atomicAdd(deg + dst[e], 1.0f);
}

// features are always D=128 wide; one wave per edge, 4 floats per lane
__global__ void scatter_add_kernel(const float* __restrict__ feat,
                                   const int* __restrict__ src, const int* __restrict__ dst,
                                   long long E, float* __restrict__ out, int ldOut, int colOff) {
  long long gid = (long long)blockIdx.x * blockDim.x + threadIdx.x;
  long long e = gid >> 5;
  if (e >= E) return;
  int lane = threadIdx.x & 31;
  int s = src[e], d = dst[e];
  const float4 v = *(const float4*)(feat + (long long)s * 128 + lane * 4);
  float* o = out + (long long)d * ldOut + colOff + lane * 4;
  atomicAdd(o + 0, v.x);
  atomicAdd(o + 1, v.y);
  atomicAdd(o + 2, v.z);
  atomicAdd(o + 3, v.w);
}

__global__ void meandiv_kernel(float* __restrict__ out, int ldOut, int colOff, int Dcols,
                               const float* __restrict__ deg, long long N) {
  long long i = (long long)blockIdx.x * blockDim.x + threadIdx.x;
  if (i >= N * (long long)Dcols) return;
  long long row = i / Dcols;
  int col = (int)(i - row * Dcols);
  float dg = fmaxf(deg[row], 1.0f);
  out[row * (long long)ldOut + colOff + col] /= dg;
}

// ---------------- hyp bias: proj(expmap0(b)) ----------------
__global__ void hypbias_kernel(const float* __restrict__ b, int M, float* __restrict__ e) {
  int lane = threadIdx.x & 31;
  float acc = 0.f;
  for (int k = lane; k < M; k += 32) { float v = b[k]; acc += v * v; }
  float n = fmaxf(sqrtf(wsum(acc)), MIN_NORM);
  float sc = tanhf(n) / n;            // expmap0
  float ne = fmaxf(fabsf(tanhf(n)), MIN_NORM);
  float ps = (ne > MAXNORM) ? (MAXNORM / ne) : 1.0f;   // proj
  for (int k = lane; k < M; k += 32) e[k] = b[k] * sc * ps;
}

// ---------------- WMMA GEMM: C[N,M] = A[N,K] @ W[M,K]^T  (f16 in, f32 out) ----------------
__global__ void wmma_gemm_f16(const _Float16* __restrict__ A, const _Float16* __restrict__ B,
                              float* __restrict__ C, int Nrows, int K, int M) {
  int wave = (int)(((long long)blockIdx.x * blockDim.x + threadIdx.x) >> 5);
  int lane = threadIdx.x & 31;
  int colTiles = M >> 4;
  int nTiles = (Nrows >> 4) * colTiles;
  if (wave >= nTiles) return;
  int ti = wave / colTiles;
  int tj = wave - ti * colTiles;
  int hi = (lane >= 16);
  int aRow = (ti << 4) + (lane & 15);
  int bRow = (tj << 4) + (lane & 15);
  // 16-bit A 16x32 fragment: lanes 0-15 hold K = k0..k0+15,
  //                          lanes 16-31 hold K = k0+16..k0+31
  const _Float16* aPtr = A + (long long)aRow * K + (hi ? 16 : 0);
  // 16-bit B 32x16 fragment: lane n = lane&15, lanes 0-15 K=k0..k0+15, lanes 16-31 K=k0+16..k0+31
  const _Float16* bPtr = B + (long long)bRow * K + (hi ? 16 : 0);
  v8f c = {};
  for (int k0 = 0; k0 < K; k0 += 32) {
    if (k0 + 32 < K) __builtin_prefetch(aPtr + k0 + 32, 0, 3);
    v8h alo = *(const v8h*)(aPtr + k0);
    v8h ahi = *(const v8h*)(aPtr + k0 + 8);
    v16h a = __builtin_shufflevector(alo, ahi, 0, 1, 2, 3, 4, 5, 6, 7,
                                     8, 9, 10, 11, 12, 13, 14, 15);
    v16h b = *(const v16h*)(bPtr + k0);
    c = __builtin_amdgcn_wmma_f32_16x16x32_f16(false, a, false, b, (short)0, c, false, false);
  }
  // D layout: VGPR r, lanes 0-15 -> M=r, lanes 16-31 -> M=8+r; N = lane&15
  int m0 = (ti << 4) + (hi ? 8 : 0);
  int n0 = (tj << 4) + (lane & 15);
  float* cp = C + (long long)m0 * M + n0;
#pragma unroll
  for (int r = 0; r < 8; ++r) cp[(long long)r * M] = c[r];
}

// ---------------- fused hyperbolic tail: matvec scale + bias mobius_add + act ----------------
__device__ __forceinline__ void proj_row(float* r, int PE) {
  float a = 0.f;
  for (int j = 0; j < PE; ++j) a += r[j] * r[j];
  float n = fmaxf(sqrtf(wsum(a)), MIN_NORM);
  if (n > MAXNORM) {
    float sc = MAXNORM / n;
    for (int j = 0; j < PE; ++j) r[j] *= sc;
  }
}

__global__ void hyp_post_kernel(const float* __restrict__ catF, int K,
                                const float* __restrict__ mx, int M,
                                const float* __restrict__ biasH,
                                float* __restrict__ out, int outStride, int outColOff,
                                long long Nrows) {
  int wavesPerBlock = blockDim.x >> 5;
  long long row = (long long)blockIdx.x * wavesPerBlock + (threadIdx.x >> 5);
  int lane = threadIdx.x & 31;
  if (row >= Nrows) return;
  // xn = ||input row||
  const float* xr = catF + row * (long long)K;
  float acc = 0.f;
  for (int k = lane; k < K; k += 32) { float v = xr[k]; acc += v * v; }
  float xn = fmaxf(sqrtf(wsum(acc)), MIN_NORM);
  // load mx row, mxn
  const float* mr = mx + row * (long long)M;
  int PE = M >> 5;                      // 4 or 8
  float r[8], e[8];
  float acc2 = 0.f;
  for (int j = 0; j < PE; ++j) { r[j] = mr[lane + 32 * j]; acc2 += r[j] * r[j]; }
  float mxn = fmaxf(sqrtf(wsum(acc2)), MIN_NORM);
  // mobius_matvec tail: tanh(mxn/xn * artanh(xn)) * mx / mxn
  float art = atanhf(fminf(xn, 1.0f - 1e-7f));
  float t = tanhf(mxn / xn * art);
  float s = t / mxn;
  for (int j = 0; j < PE; ++j) r[j] *= s;
  proj_row(r, PE);
  // mobius_add with hyperbolic bias
  float x2a = 0.f, y2a = 0.f, xya = 0.f;
  for (int j = 0; j < PE; ++j) {
    e[j] = biasH[lane + 32 * j];
    x2a += r[j] * r[j];
    y2a += e[j] * e[j];
    xya += r[j] * e[j];
  }
  float x2 = wsum(x2a), y2 = wsum(y2a), xy = wsum(xya);
  float den = fmaxf(1.f + 2.f * xy + x2 * y2, MIN_NORM);
  float ca = (1.f + 2.f * xy + y2) / den;
  float cb = (1.f - x2) / den;
  for (int j = 0; j < PE; ++j) r[j] = ca * r[j] + cb * e[j];
  proj_row(r, PE);
  // hyp_act: relu(logmap0) -> expmap0 -> proj
  float n2 = 0.f;
  for (int j = 0; j < PE; ++j) n2 += r[j] * r[j];
  float n = fmaxf(sqrtf(wsum(n2)), MIN_NORM);
  float sl = atanhf(fminf(n, 1.0f - 1e-7f)) / n;
  for (int j = 0; j < PE; ++j) r[j] = fmaxf(r[j] * sl, 0.0f);
  float u2 = 0.f;
  for (int j = 0; j < PE; ++j) u2 += r[j] * r[j];
  float nu = fmaxf(sqrtf(wsum(u2)), MIN_NORM);
  float se = tanhf(nu) / nu;
  for (int j = 0; j < PE; ++j) r[j] *= se;
  proj_row(r, PE);
  float* orow = out + row * (long long)outStride + outColOff;
  for (int j = 0; j < PE; ++j) orow[lane + 32 * j] = r[j];
}

// ---------------- host side ----------------
static inline unsigned gblocks(long long threads, int bs) {
  return (unsigned)((threads + bs - 1) / bs);
}

extern "C" void kernel_launch(void* const* d_in, const int* in_sizes, int n_in,
                              void* d_out, int out_size, void* d_ws, size_t ws_size,
                              hipStream_t stream) {
  const int D = 128;
  const long long N = in_sizes[0] / D;
  const long long E = in_sizes[1];

  const float* x       = (const float*)d_in[0];
  const int*   src_pos = (const int*)d_in[1];
  const int*   dst_pos = (const int*)d_in[2];
  const int*   src_neg = (const int*)d_in[3];
  const int*   dst_neg = (const int*)d_in[4];
  const float* W1b = (const float*)d_in[5];  const float* b1b = (const float*)d_in[6];
  const float* W1h = (const float*)d_in[7];  const float* b1h = (const float*)d_in[8];
  const float* W2b = (const float*)d_in[9];  const float* b2b = (const float*)d_in[10];
  const float* W2h = (const float*)d_in[11]; const float* b2h = (const float*)d_in[12];
  const float* W3b = (const float*)d_in[13]; const float* b3b = (const float*)d_in[14];
  const float* W3h = (const float*)d_in[15]; const float* b3h = (const float*)d_in[16];
  float* out = (float*)d_out;

  // workspace bump allocator (256B aligned)
  char* ws = (char*)d_ws;
  size_t off = 0;
  auto alloc = [&](size_t bytes) -> void* {
    void* p = ws + off;
    off += (bytes + 255) & ~(size_t)255;
    return p;
  };
  float* deg_pos = (float*)alloc(N * 4);
  float* deg_neg = (float*)alloc(N * 4);
  float* pos1 = (float*)alloc(N * 128 * 4);
  float* neg1 = (float*)alloc(N * 128 * 4);
  float* hb1  = (float*)alloc(N * 128 * 4);
  float* hn1  = (float*)alloc(N * 128 * 4);
  float* pos2 = (float*)alloc(N * 256 * 4);
  float* neg2 = (float*)alloc(N * 256 * 4);
  float* hb2  = (float*)alloc(N * 256 * 4);
  float* hn2  = (float*)alloc(N * 256 * 4);
  float* cat  = (float*)alloc(N * 384 * 4);
  float* mxb  = (float*)alloc(N * 256 * 4);
  _Float16* catH = (_Float16*)alloc(N * 384 * 2);
  _Float16* wH   = (_Float16*)alloc(384 * 256 * 2);
  float* biasH   = (float*)alloc(384 * 4);

  const int BS = 256;
  auto run_layer = [&](const float* catF, int K, const float* W, const float* bv, int M,
                       float* outPtr, int outStride, int outColOff) {
    cvt_f16_kernel<<<gblocks(N * K, BS), BS, 0, stream>>>(catF, catH, N * (long long)K);
    cvt_f16_kernel<<<gblocks((long long)M * K, BS), BS, 0, stream>>>(W, wH, (long long)M * K);
    hypbias_kernel<<<1, 32, 0, stream>>>(bv, M, biasH);
    long long nT = (N / 16) * (M / 16);
    wmma_gemm_f16<<<gblocks(nT * 32, BS), BS, 0, stream>>>(catH, wH, mxb, (int)N, K, M);
    hyp_post_kernel<<<gblocks(N * 32, 128), 128, 0, stream>>>(catF, K, mxb, M, biasH,
                                                              outPtr, outStride, outColOff, N);
  };

  // degrees
  zero_kernel<<<gblocks(N, BS), BS, 0, stream>>>(deg_pos, N);
  zero_kernel<<<gblocks(N, BS), BS, 0, stream>>>(deg_neg, N);
  degree_kernel<<<gblocks(E, BS), BS, 0, stream>>>(dst_pos, E, deg_pos);
  degree_kernel<<<gblocks(E, BS), BS, 0, stream>>>(dst_neg, E, deg_neg);

  // ---- layer 1 aggregations ----
  zero_kernel<<<gblocks(N * 128, BS), BS, 0, stream>>>(pos1, N * 128);
  zero_kernel<<<gblocks(N * 128, BS), BS, 0, stream>>>(neg1, N * 128);
  scatter_add_kernel<<<gblocks(E * 32, BS), BS, 0, stream>>>(x, src_pos, dst_pos, E, pos1, 128, 0);
  scatter_add_kernel<<<gblocks(E * 32, BS), BS, 0, stream>>>(x, src_neg, dst_neg, E, neg1, 128, 0);
  meandiv_kernel<<<gblocks(N * 128, BS), BS, 0, stream>>>(pos1, 128, 0, 128, deg_pos, N);
  meandiv_kernel<<<gblocks(N * 128, BS), BS, 0, stream>>>(neg1, 128, 0, 128, deg_neg, N);

  // ---- layer 1 hyperbolic linear + act ----
  concat_kernel<<<gblocks(N * 256, BS), BS, 0, stream>>>(pos1, 128, x, 128, cat, N);
  run_layer(cat, 256, W1b, b1b, 128, hb1, 128, 0);
  concat_kernel<<<gblocks(N * 256, BS), BS, 0, stream>>>(neg1, 128, x, 128, cat, N);
  run_layer(cat, 256, W1h, b1h, 128, hn1, 128, 0);

  // ---- layer 2 balance-theory aggregations ----
  zero_kernel<<<gblocks(N * 256, BS), BS, 0, stream>>>(pos2, N * 256);
  zero_kernel<<<gblocks(N * 256, BS), BS, 0, stream>>>(neg2, N * 256);
  scatter_add_kernel<<<gblocks(E * 32, BS), BS, 0, stream>>>(hb1, src_pos, dst_pos, E, pos2, 256, 0);
  scatter_add_kernel<<<gblocks(E * 32, BS), BS, 0, stream>>>(hn1, src_neg, dst_neg, E, pos2, 256, 128);
  scatter_add_kernel<<<gblocks(E * 32, BS), BS, 0, stream>>>(hn1, src_pos, dst_pos, E, neg2, 256, 0);
  scatter_add_kernel<<<gblocks(E * 32, BS), BS, 0, stream>>>(hb1, src_neg, dst_neg, E, neg2, 256, 128);
  meandiv_kernel<<<gblocks(N * 128, BS), BS, 0, stream>>>(pos2, 256, 0, 128, deg_pos, N);
  meandiv_kernel<<<gblocks(N * 128, BS), BS, 0, stream>>>(pos2, 256, 128, 128, deg_neg, N);
  meandiv_kernel<<<gblocks(N * 128, BS), BS, 0, stream>>>(neg2, 256, 0, 128, deg_pos, N);
  meandiv_kernel<<<gblocks(N * 128, BS), BS, 0, stream>>>(neg2, 256, 128, 128, deg_neg, N);

  // ---- layer 2 hyperbolic linear + act ----
  concat_kernel<<<gblocks(N * 384, BS), BS, 0, stream>>>(pos2, 256, hb1, 128, cat, N);
  run_layer(cat, 384, W2b, b2b, 256, hb2, 256, 0);
  concat_kernel<<<gblocks(N * 384, BS), BS, 0, stream>>>(neg2, 256, hn1, 128, cat, N);
  run_layer(cat, 384, W2h, b2h, 256, hn2, 256, 0);

  // ---- layer 3 (no aggregation) -> directly into d_out [N, 384] ----
  run_layer(hb2, 256, W3b, b3b, 128, out, 384, 0);
  run_layer(hn2, 256, W3h, b3h, 128, out, 384, 128);
  copyx_kernel<<<gblocks(N * 128, BS), BS, 0, stream>>>(x, out, N);
}
